// ConLoss_75926431859348
// MI455X (gfx1250) — matrix operand order repaired and checked by reading, not compiled, not run on hardware
//
#include <hip/hip_runtime.h>
#include <hip/hip_bf16.h>

typedef __attribute__((ext_vector_type(16))) __bf16 v16bf;
typedef __attribute__((ext_vector_type(8)))  __bf16 v8bf;
typedef __attribute__((ext_vector_type(4)))  __bf16 v4bf;
typedef __attribute__((ext_vector_type(8)))  float  v8f;

constexpr int   kN      = 4096;
constexpr int   kD      = 512;
constexpr int   kCls    = 100;
constexpr float kInvTau = 10.0f;   // 1 / 0.1

// ---------------------------------------------------------------------------
// Kernel 1: class histogram -> stats[0..99] = count per class (float),
//           stats[100] = sum of squared counts (= sum_i P_i).
// ---------------------------------------------------------------------------
__global__ void conloss_stats_kernel(const int* __restrict__ targets,
                                     float* __restrict__ stats) {
    __shared__ int hist[kCls];
    const int tid = threadIdx.x;
    for (int c = tid; c < kCls; c += blockDim.x) hist[c] = 0;
    __syncthreads();
    for (int i = tid; i < kN; i += blockDim.x) atomicAdd(&hist[targets[i]], 1);
    __syncthreads();
    for (int c = tid; c < kCls; c += blockDim.x) stats[c] = (float)hist[c];
    if (tid == 0) {
        float sp = 0.f;
        for (int c = 0; c < kCls; ++c) sp += (float)hist[c] * (float)hist[c];
        stats[kCls] = sp;
    }
}

// ---------------------------------------------------------------------------
// Kernel 2: row-normalize fp32 features (cosine prep) and emit bf16 matrix G.
// One block (128 threads = 4 waves) per row; each thread handles 4 elements.
// ---------------------------------------------------------------------------
__global__ void conloss_norm_kernel(const float* __restrict__ F,
                                    __bf16* __restrict__ G) {
    const int row = blockIdx.x;
    const int tid = threadIdx.x;  // 128
    const float4 v = ((const float4*)(F + (size_t)row * kD))[tid];
    float ss = v.x * v.x + v.y * v.y + v.z * v.z + v.w * v.w;
#pragma unroll
    for (int off = 16; off; off >>= 1) ss += __shfl_xor(ss, off);
    __shared__ float red[4];
    if ((tid & 31) == 0) red[tid >> 5] = ss;
    __syncthreads();
    const float tot = red[0] + red[1] + red[2] + red[3];
    // zero-norm guard matches reference (0/1e-8 == 0)
    const float rn = tot > 0.f ? 1.0f / sqrtf(tot) : 0.f;
    v4bf o;
    o[0] = (__bf16)(v.x * rn);
    o[1] = (__bf16)(v.y * rn);
    o[2] = (__bf16)(v.z * rn);
    o[3] = (__bf16)(v.w * rn);
    ((v4bf*)(G + (size_t)row * kD))[tid] = o;
}

// ---------------------------------------------------------------------------
// Kernel 3: fused WMMA GEMM + contrastive loss.
// Block = 16 rows x 4096 cols, 256 threads (8 waves, 512 cols each).
// Dynamic LDS: sim block (16 x 4096 bf16 = 128KB) + targets (16KB).
// ---------------------------------------------------------------------------
__global__ __launch_bounds__(256) void conloss_main_kernel(
        const __bf16* __restrict__ G, const int* __restrict__ targets,
        const float* __restrict__ stats, float* __restrict__ partial) {
    extern __shared__ char smem[];
    __bf16* sim_sh = (__bf16*)smem;                                   // 16*4096 bf16
    int*    tgt_sh = (int*)(smem + (size_t)16 * kN * sizeof(__bf16)); // 4096 ints
    __shared__ float negsum[16];
    __shared__ float wgsum[8];

    const int tid   = threadIdx.x;
    const int lane  = tid & 31;
    const int w     = tid >> 5;
    const int half  = lane >> 4;   // 0 or 1
    const int m     = lane & 15;   // row (A) / col (B) within tile
    const int r0    = blockIdx.x * 16;
    const int wbase = w * 512;

    for (int i = tid; i < kN; i += 256) tgt_sh[i] = targets[i];
    if (tid < 16) negsum[tid] = 0.f;
    __syncthreads();

    // --- preload A fragments (16 block rows, all K) -----------------------
    // 16-bit A 16x32 layout: lanes 0-15 -> M=lane, elems 0..7 = K 0..7 (half 0)
    // / K 8..15 (half 1); elems 8..15 = K 16..23 / 24..31.
    const __bf16* Arow = G + (size_t)(r0 + m) * kD;
    v16bf afrag[16];
#pragma unroll
    for (int kk = 0; kk < 16; ++kk) {
        const v8bf lo = *(const v8bf*)(Arow + kk * 32 + half * 8);
        const v8bf hi = *(const v8bf*)(Arow + kk * 32 + 16 + half * 8);
#pragma unroll
        for (int e = 0; e < 8; ++e) { afrag[kk][e] = lo[e]; afrag[kk][8 + e] = hi[e]; }
    }
    int trow[8];
#pragma unroll
    for (int r = 0; r < 8; ++r) trow[r] = tgt_sh[r0 + r + 8 * half];

    // --- phase 1: GEMM tiles, exp, negative sums, stage sims in LDS -------
    float negpart[8] = {0.f, 0.f, 0.f, 0.f, 0.f, 0.f, 0.f, 0.f};
    for (int t = 0; t < 32; ++t) {
        const int col0 = wbase + t * 16;
        // B 32x16 layout: lane -> N=m; elems 0..7 = K half*8..half*8+7, elems 8..15 = K 16+half*8..23+half*8
        const __bf16* Brow = G + (size_t)(col0 + m) * kD;
        v8f acc = {};
#pragma unroll
        for (int kk = 0; kk < 16; ++kk) {
            const v8bf blo = *(const v8bf*)(Brow + kk * 32 + half * 8);
            const v8bf bhi = *(const v8bf*)(Brow + kk * 32 + 16 + half * 8);
            v16bf b;
#pragma unroll
            for (int e = 0; e < 8; ++e) { b[e] = blo[e]; b[8 + e] = bhi[e]; }
            acc = __builtin_amdgcn_wmma_f32_16x16x32_bf16(
                false, afrag[kk], false, b, (short)0, acc, false, false);
        }
        const int gj = col0 + m;
        const int tj = tgt_sh[gj];
#pragma unroll
        for (int r = 0; r < 8; ++r) {
            const int   R = r + 8 * half;        // D layout: VGPR r -> M = r+8*half
            const float s = acc[r];
            sim_sh[(size_t)R * kN + gj] = (__bf16)s;
            if (tj != trow[r]) negpart[r] += __expf(s * kInvTau);  // diag is same-class
        }
    }
#pragma unroll
    for (int r = 0; r < 8; ++r) {
        float v = negpart[r];
        v += __shfl_xor(v, 1); v += __shfl_xor(v, 2);
        v += __shfl_xor(v, 4); v += __shfl_xor(v, 8);
        if (m == 0) atomicAdd(&negsum[r + 8 * half], v);
    }
    __syncthreads();

    // --- phase 2: positive pairs: log(e_ij + neg_i) - s_ij/tau ------------
    float accl = 0.f;
    for (int R = 0; R < 16; ++R) {
        const int   gi  = r0 + R;
        const int   ti  = tgt_sh[gi];
        const float ns  = negsum[R];
        const float ipc = 1.0f / stats[ti];      // 1 / positive_cardinality
        for (int j = wbase + lane; j < wbase + 512; j += 32) {
            if (tgt_sh[j] == ti && j != gi) {
                const float st = (float)sim_sh[(size_t)R * kN + j] * kInvTau;
                accl += (__logf(__expf(st) + ns) - st) * ipc;
            }
        }
    }
    accl += __shfl_xor(accl, 1);  accl += __shfl_xor(accl, 2);
    accl += __shfl_xor(accl, 4);  accl += __shfl_xor(accl, 8);
    accl += __shfl_xor(accl, 16);
    if (lane == 0) wgsum[w] = accl;
    __syncthreads();
    if (tid == 0) {
        float tot = 0.f;
#pragma unroll
        for (int i = 0; i < 8; ++i) tot += wgsum[i];
        partial[blockIdx.x] = tot;
    }
}

// ---------------------------------------------------------------------------
// Kernel 4: deterministic final reduction of 256 block partials.
// ---------------------------------------------------------------------------
__global__ void conloss_final_kernel(const float* __restrict__ partial,
                                     const float* __restrict__ stats,
                                     float* __restrict__ out) {
    __shared__ float red[256];
    const int tid = threadIdx.x;
    red[tid] = partial[tid];
    __syncthreads();
    for (int s = 128; s; s >>= 1) {
        if (tid < s) red[tid] += red[tid + s];
        __syncthreads();
    }
    if (tid == 0) out[0] = red[0] / stats[kCls];   // divide by sum of P_i
}

// ---------------------------------------------------------------------------
extern "C" void kernel_launch(void* const* d_in, const int* in_sizes, int n_in,
                              void* d_out, int out_size, void* d_ws, size_t ws_size,
                              hipStream_t stream) {
    (void)in_sizes; (void)n_in; (void)out_size; (void)ws_size;
    const float* features = (const float*)d_in[0];
    const int*   targets  = (const int*)d_in[1];
    float*       out      = (float*)d_out;

    // workspace layout: [G bf16: 4096*512*2 = 4MB][stats: 128 floats][partial: 256 floats]
    __bf16* G       = (__bf16*)d_ws;
    float*  stats   = (float*)((char*)d_ws + (size_t)kN * kD * sizeof(__bf16));
    float*  partial = stats + 128;

    conloss_stats_kernel<<<1, 256, 0, stream>>>(targets, stats);
    conloss_norm_kernel<<<kN, 128, 0, stream>>>(features, G);

    const size_t smem = (size_t)16 * kN * sizeof(__bf16) + (size_t)kN * sizeof(int);
    conloss_main_kernel<<<kN / 16, 256, smem, stream>>>(G, targets, stats, partial);
    conloss_final_kernel<<<1, 256, 0, stream>>>(partial, stats, out);
}
